// MultiHeadSelfAttention_25769804120
// MI455X (gfx1250) — matrix operand order repaired and checked by reading, not compiled, not run on hardware
//
#include <hip/hip_runtime.h>

#define BS   4
#define SEQ  2048
#define DIM  768
#define NH   12
#define DH   64
#define MTOT (BS * SEQ)   // 8192
#define NIT  (SEQ / 32)   // 64 key steps

typedef __bf16 v16bf __attribute__((ext_vector_type(16)));
typedef __bf16 v8bf  __attribute__((ext_vector_type(8)));
typedef float  v8f   __attribute__((ext_vector_type(8)));
typedef unsigned int u32x4 __attribute__((ext_vector_type(4)));
typedef int          i32x4 __attribute__((ext_vector_type(4)));
typedef int          i32x8 __attribute__((ext_vector_type(8)));

#if defined(__has_builtin)
#if __has_builtin(__builtin_amdgcn_tensor_load_to_lds)
#define HAVE_TDM 1
#endif
#endif
#ifndef HAVE_TDM
#define HAVE_TDM 0
#endif

__device__ __forceinline__ v8f wmma_bf16(v16bf a, v16bf b, v8f c) {
  return __builtin_amdgcn_wmma_f32_16x16x32_bf16(
      /*neg_a=*/false, a, /*neg_b=*/false, b,
      /*c_mod=*/(short)0, c, /*reuse_a=*/false, /*reuse_b=*/false);
}

#if HAVE_TDM
// 2D TDM tile load Global->LDS, bf16 elements, with LDS row padding.
__device__ __forceinline__ void tdm_load_2d(unsigned lds_off, const void* gptr,
                                            unsigned tile_x, unsigned tile_y,
                                            unsigned tensor_x, unsigned tensor_y,
                                            unsigned stride_x,
                                            unsigned pad_int_c, unsigned pad_amt_c) {
  unsigned long long ga = (unsigned long long)gptr;
  u32x4 g0;
  g0[0] = 1u;                                           // count=1 (valid D#)
  g0[1] = lds_off;                                      // lds_addr
  g0[2] = (unsigned)(ga & 0xFFFFFFFFu);                 // global_addr[31:0]
  g0[3] = (unsigned)((ga >> 32) & 0x01FFFFFFu) | (2u << 30);  // ga[56:32] | type=2
  i32x8 g1;
  g1[0] = (int)((1u << 16)            // data_size = 2 bytes
              | (1u << 20)            // pad_enable
              | (pad_int_c << 22)
              | (pad_amt_c << 25));
  g1[1] = (int)((tensor_x & 0xFFFFu) << 16);
  g1[2] = (int)((tensor_x >> 16) | ((tensor_y & 0xFFFFu) << 16));
  g1[3] = (int)((tensor_y >> 16) | (tile_x << 16));
  g1[4] = (int)(tile_y & 0xFFFFu);
  g1[5] = (int)stride_x;
  g1[6] = 0;
  g1[7] = 0;
  i32x4 gz = {0, 0, 0, 0};
#if __clang_major__ >= 23
  i32x8 gz8 = {0, 0, 0, 0, 0, 0, 0, 0};
  __builtin_amdgcn_tensor_load_to_lds(g0, g1, gz, gz, gz8, 0);
#else
  __builtin_amdgcn_tensor_load_to_lds(g0, g1, gz, gz, 0);
#endif
}
#endif

// ---------------------------------------------------------------------------
// Projection GEMM: out = (A[M,K] @ W[N,K]^T + bias) * scale, emitted as bf16.
// vtrans==0 -> out laid out [b][h][s][d]; vtrans==1 -> [b][h][d][s].
// ---------------------------------------------------------------------------
__global__ __launch_bounds__(128)
void proj_kernel(const float* __restrict__ A, const float* __restrict__ W,
                 const float* __restrict__ bias, __bf16* __restrict__ out,
                 float scale, int vtrans) {
  const int lane = threadIdx.x & 31;
  const int wave = threadIdx.x >> 5;
  const int ln   = lane & 15;
  const int hl   = lane >> 4;
  const int m0 = blockIdx.x * 64 + (wave >> 1) * 32;
  const int n0 = blockIdx.y * 64 + (wave & 1) * 32;

  v8f c[2][2] = {};
  for (int kk = 0; kk < DIM; kk += 32) {
    v16bf a[2], b[2];
#pragma unroll
    for (int i = 0; i < 2; i++) {
      const float* ap = A + (size_t)(m0 + i * 16 + ln) * DIM + kk + hl * 8;
#pragma unroll
      for (int t = 0; t < 8; t++) {
        a[i][t]     = (__bf16)ap[t];
        a[i][8 + t] = (__bf16)ap[16 + t];
      }
    }
#pragma unroll
    for (int j = 0; j < 2; j++) {
      const float* wp = W + (size_t)(n0 + j * 16 + ln) * DIM + kk + hl * 8;
#pragma unroll
      for (int t = 0; t < 8; t++) { b[j][t] = (__bf16)wp[t]; b[j][8 + t] = (__bf16)wp[16 + t]; }
    }
#pragma unroll
    for (int i = 0; i < 2; i++)
#pragma unroll
      for (int j = 0; j < 2; j++) c[i][j] = wmma_bf16(a[i], b[j], c[i][j]);
  }

#pragma unroll
  for (int i = 0; i < 2; i++)
#pragma unroll
    for (int j = 0; j < 2; j++) {
      const int n_g = n0 + j * 16 + ln;
      const float bv = bias[n_g];
      const int h = n_g >> 6, d = n_g & 63;
#pragma unroll
      for (int r = 0; r < 8; r++) {
        const int m_g = m0 + i * 16 + r + hl * 8;
        const int bi = m_g >> 11, s = m_g & (SEQ - 1);
        const float val = (c[i][j][r] + bv) * scale;
        const size_t idx = vtrans
            ? ((size_t)((bi * NH + h) * DH + d) * SEQ + s)
            : ((size_t)((bi * NH + h) * SEQ + s) * DH + d);
        out[idx] = (__bf16)val;
      }
    }
}

// ---------------------------------------------------------------------------
// Flash attention. Block = 8 waves, all on one (b,h), 128 query rows.
// K/V tiles double-buffered in LDS via async TDM: the DMA for step i+1 is in
// flight while step i computes; s_wait_tensorcnt(2) relies on per-wave
// in-order TDM completion to release the older tile pair.
// ---------------------------------------------------------------------------
#define KPITCH 72   // 64 + 8 bf16 pad  (row 128B + 16B pad)
#define VPITCH 40   // 32 + 8 bf16 pad  (row 64B + 16B pad)

__global__ __launch_bounds__(256)
void attn_kernel(const __bf16* __restrict__ qb, const __bf16* __restrict__ kb,
                 const __bf16* __restrict__ vt, const int* __restrict__ mask,
                 __bf16* __restrict__ ctx) {
  __shared__ __align__(16) __bf16 ktile[2][32][KPITCH];   // 9 KB
  __shared__ __align__(16) __bf16 vtile[2][64][VPITCH];   // 10 KB
  __shared__ __align__(16) __bf16 ptile[8][16][32];       // 8 KB

  const int lane = threadIdx.x & 31;
  const int wv   = threadIdx.x >> 5;
  const int ln   = lane & 15;
  const int hl   = lane >> 4;
  const int bh   = blockIdx.x >> 4;        // 16 q-blocks per (b,h)
  const int qblk = blockIdx.x & 15;
  const int b  = bh / NH;
  const int h  = bh % NH;
  const int q0 = qblk * 128 + wv * 16;     // this wave's 16 query rows

  // Q A-frags (Q already scaled by 1/8, bf16)
  v16bf qa[2];
  const __bf16* qp = qb + ((size_t)bh * SEQ + q0 + ln) * DH + hl * 8;
#pragma unroll
  for (int dk = 0; dk < 2; dk++) {
    v8bf c0 = *(const v8bf*)(qp + dk * 32);
    v8bf c1 = *(const v8bf*)(qp + dk * 32 + 16);
#pragma unroll
    for (int t = 0; t < 8; t++) { qa[dk][t] = c0[t]; qa[dk][8 + t] = c1[t]; }
  }

  v8f o[4] = {};
  float mrow[8], lrow[8];
#pragma unroll
  for (int r = 0; r < 8; r++) { mrow[r] = -3.0e38f; lrow[r] = 0.0f; }

  const __bf16* kp0 = kb + (size_t)bh * SEQ * DH;
  const __bf16* vp0 = vt + (size_t)bh * DH * SEQ;
  const int* mp = mask + b * SEQ;
  const float NEG = -3.0e38f;

#if HAVE_TDM
  if (wv == 0) {  // prologue: prefetch tile 0 into buffer 0
    tdm_load_2d((unsigned)(size_t)&ktile[0][0][0], kp0,
                DH, 32, DH, SEQ, DH, 4u, 3u);
    tdm_load_2d((unsigned)(size_t)&vtile[0][0][0], vp0,
                32, DH, SEQ, DH, SEQ, 3u, 3u);
  }
#endif

  for (int it = 0; it < NIT; it++) {
    const int k0 = it * 32;
    const int pb = it & 1;
#if HAVE_TDM
    if (wv == 0) {
      if (it + 1 < NIT) {     // issue next tile pair into the other buffer
        const int nk0 = k0 + 32, nb = pb ^ 1;
        tdm_load_2d((unsigned)(size_t)&ktile[nb][0][0], kp0 + (size_t)nk0 * DH,
                    DH, 32, DH, SEQ, DH, 4u, 3u);
        tdm_load_2d((unsigned)(size_t)&vtile[nb][0][0], vp0 + nk0,
                    32, DH, SEQ, DH, SEQ, 3u, 3u);
        __builtin_amdgcn_s_wait_tensorcnt(2);   // older pair (this tile) done
      } else {
        __builtin_amdgcn_s_wait_tensorcnt(0);
      }
    }
#else
    {  // cooperative fallback into buffer pb: 256 threads x 16B chunks
      const int t = threadIdx.x;
      {  // K tile: 32 rows x 64 elems = 8 chunks/row
        const int r = t >> 3, cch = (t & 7) * 8;
        *(v8bf*)&ktile[pb][r][cch] =
            *(const v8bf*)(kp0 + (size_t)(k0 + r) * DH + cch);
      }
      {  // V tile: 64 rows x 32 elems = 4 chunks/row
        const int r = t >> 2, cch = (t & 3) * 8;
        *(v8bf*)&vtile[pb][r][cch] =
            *(const v8bf*)(vp0 + (size_t)r * SEQ + k0 + cch);
      }
    }
#endif
    __syncthreads();   // tile pb ready & visible to all waves

    // ---- scores S = Q x K^T from LDS ----
    v8f s[2] = {{}, {}};
#pragma unroll
    for (int t = 0; t < 2; t++) {
#pragma unroll
      for (int dk = 0; dk < 2; dk++) {
        const __bf16* kp = &ktile[pb][t * 16 + ln][dk * 32 + hl * 8];
        v8bf c0 = *(const v8bf*)kp;
        v8bf c1 = *(const v8bf*)(kp + 16);
        v16bf bf;
#pragma unroll
        for (int e = 0; e < 8; e++) { bf[e] = c0[e]; bf[8 + e] = c1[e]; }
        s[t] = wmma_bf16(qa[dk], bf, s[t]);
      }
    }

    // ---- key padding mask ----
    const int mk0 = mp[k0 + ln];
    const int mk1 = mp[k0 + 16 + ln];
#pragma unroll
    for (int r = 0; r < 8; r++) {
      if (!mk0) s[0][r] = NEG;
      if (!mk1) s[1][r] = NEG;
    }

    // ---- online softmax (row reductions inside 16-lane halves) ----
#pragma unroll
    for (int r = 0; r < 8; r++) {
      float v = fmaxf(s[0][r], s[1][r]);
      v = fmaxf(v, __shfl_xor(v, 1, 32));
      v = fmaxf(v, __shfl_xor(v, 2, 32));
      v = fmaxf(v, __shfl_xor(v, 4, 32));
      v = fmaxf(v, __shfl_xor(v, 8, 32));
      const float mn = fmaxf(mrow[r], v);
      const float corr = __expf(mrow[r] - mn);
      mrow[r] = mn;
      const float p0 = __expf(s[0][r] - mn);
      const float p1 = __expf(s[1][r] - mn);
      float rs = p0 + p1;
      rs += __shfl_xor(rs, 1, 32);
      rs += __shfl_xor(rs, 2, 32);
      rs += __shfl_xor(rs, 4, 32);
      rs += __shfl_xor(rs, 8, 32);
      lrow[r] = lrow[r] * corr + rs;
#pragma unroll
      for (int j = 0; j < 4; j++) o[j][r] *= corr;
      ptile[wv][r + hl * 8][ln]      = (__bf16)p0;
      ptile[wv][r + hl * 8][16 + ln] = (__bf16)p1;
    }

    asm volatile("s_wait_dscnt 0" ::: "memory");  // wave-local LDS RAW fence

    // ---- reload P in A-matrix layout ----
    v16bf pf;
    {
      const __bf16* pp = &ptile[wv][ln][hl * 8];
      v8bf c0 = *(const v8bf*)pp;
      v8bf c1 = *(const v8bf*)(pp + 16);
#pragma unroll
      for (int e = 0; e < 8; e++) { pf[e] = c0[e]; pf[8 + e] = c1[e]; }
    }

    // ---- O += P x V from LDS ----
#pragma unroll
    for (int j = 0; j < 4; j++) {
      const __bf16* vp = &vtile[pb][j * 16 + ln][hl * 8];
      v8bf c0 = *(const v8bf*)vp;
      v8bf c1 = *(const v8bf*)(vp + 16);
      v16bf bf;
#pragma unroll
      for (int e = 0; e < 8; e++) { bf[e] = c0[e]; bf[8 + e] = c1[e]; }
      o[j] = wmma_bf16(pf, bf, o[j]);
    }

    __syncthreads();   // readers of tile pb done before it is overwritten
  }

  // ---- normalize and write context bf16 [b][s][h*64+d] ----
#pragma unroll
  for (int r = 0; r < 8; r++) {
    const float inv = 1.0f / lrow[r];
    const int q = q0 + r + hl * 8;
    __bf16* cp = ctx + ((size_t)b * SEQ + q) * DIM + h * DH;
#pragma unroll
    for (int j = 0; j < 4; j++) cp[j * 16 + ln] = (__bf16)(o[j][r] * inv);
  }
}

// ---------------------------------------------------------------------------
// Output projection: out_f32 = ctx_bf16 @ o_w^T + o_b
// ---------------------------------------------------------------------------
__global__ __launch_bounds__(128)
void outproj_kernel(const __bf16* __restrict__ A, const float* __restrict__ W,
                    const float* __restrict__ bias, float* __restrict__ out) {
  const int lane = threadIdx.x & 31;
  const int wave = threadIdx.x >> 5;
  const int ln   = lane & 15;
  const int hl   = lane >> 4;
  const int m0 = blockIdx.x * 64 + (wave >> 1) * 32;
  const int n0 = blockIdx.y * 64 + (wave & 1) * 32;

  v8f c[2][2] = {};
  for (int kk = 0; kk < DIM; kk += 32) {
    v16bf a[2], b[2];
#pragma unroll
    for (int i = 0; i < 2; i++) {
      const __bf16* ap = A + (size_t)(m0 + i * 16 + ln) * DIM + kk + hl * 8;
      v8bf c0 = *(const v8bf*)ap;
      v8bf c1 = *(const v8bf*)(ap + 16);
#pragma unroll
      for (int t = 0; t < 8; t++) { a[i][t] = c0[t]; a[i][8 + t] = c1[t]; }
    }
#pragma unroll
    for (int j = 0; j < 2; j++) {
      const float* wp = W + (size_t)(n0 + j * 16 + ln) * DIM + kk + hl * 8;
#pragma unroll
      for (int t = 0; t < 8; t++) { b[j][t] = (__bf16)wp[t]; b[j][8 + t] = (__bf16)wp[16 + t]; }
    }
#pragma unroll
    for (int i = 0; i < 2; i++)
#pragma unroll
      for (int j = 0; j < 2; j++) c[i][j] = wmma_bf16(a[i], b[j], c[i][j]);
  }

#pragma unroll
  for (int i = 0; i < 2; i++)
#pragma unroll
    for (int j = 0; j < 2; j++) {
      const int n_g = n0 + j * 16 + ln;
      const float bv = bias[n_g];
#pragma unroll
      for (int r = 0; r < 8; r++) {
        const int m_g = m0 + i * 16 + r + hl * 8;
        out[(size_t)m_g * DIM + n_g] = c[i][j][r] + bv;
      }
    }
}

// ---------------------------------------------------------------------------
extern "C" void kernel_launch(void* const* d_in, const int* in_sizes, int n_in,
                              void* d_out, int out_size, void* d_ws, size_t ws_size,
                              hipStream_t stream) {
  const float* query = (const float*)d_in[0];
  const float* key_  = (const float*)d_in[1];
  const float* value = (const float*)d_in[2];
  const int*   mask  = (const int*)d_in[3];
  const float* q_w = (const float*)d_in[4];
  const float* q_b = (const float*)d_in[5];
  const float* k_w = (const float*)d_in[6];
  const float* k_b = (const float*)d_in[7];
  const float* v_w = (const float*)d_in[8];
  const float* v_b = (const float*)d_in[9];
  const float* o_w = (const float*)d_in[10];
  const float* o_b = (const float*)d_in[11];

  const size_t QE = (size_t)BS * NH * SEQ * DH;
  __bf16* qbf = (__bf16*)d_ws;
  __bf16* kbf = qbf + QE;
  __bf16* vtb = kbf + QE;
  __bf16* ctx = vtb + QE;

  dim3 gproj(MTOT / 64, DIM / 64);  // (128, 12)
  proj_kernel<<<gproj, 128, 0, stream>>>(query, q_w, q_b, qbf, 0.125f, 0);
  proj_kernel<<<gproj, 128, 0, stream>>>(key_,  k_w, k_b, kbf, 1.0f,   0);
  proj_kernel<<<gproj, 128, 0, stream>>>(value, v_w, v_b, vtb, 1.0f,   1);

  const int nblk = BS * NH * (SEQ / 128);  // 768 blocks, 128 q-rows each
  attn_kernel<<<nblk, 256, 0, stream>>>(qbf, kbf, vtb, mask, ctx);

  outproj_kernel<<<gproj, 128, 0, stream>>>(ctx, o_w, o_b, (float*)d_out);
}
